// GraphAttentionLayer_8546984919264
// MI455X (gfx1250) — matrix operand order repaired and checked by reading, not compiled, not run on hardware
//
#include <hip/hip_runtime.h>
#include <hip/hip_bf16.h>

// Problem constants (from reference)
#define BB   16
#define LL   32
#define AA   2046
#define MM   2048    // = AA + 2
#define NN6L 192     // 6*L rows of raw

typedef __attribute__((ext_vector_type(16))) _Float16 v16h;
typedef __attribute__((ext_vector_type(8)))  float    v8f;

// raw[b, n, m]: col 0 = user, cols 1..2046 = attributor*W_1, col 2047 = item
__device__ __forceinline__ float raw_val(const float* __restrict__ user,
                                         const float* __restrict__ item,
                                         const float* __restrict__ att,
                                         const float* __restrict__ W1,
                                         int b, int n, int m) {
    if (m == 0)        return user[b * NN6L + n];
    if (m == MM - 1)   return item[b * NN6L + n];
    const int c = m - 1;
    return att[(size_t)b * NN6L * AA + (size_t)n * AA + c] * W1[n * AA + c];
}

// ---------------------------------------------------------------------------
// Kernel 1: Q/K/V via WMMA.
//   out_s[l, m] = sum_{n=0..63} relu(raw[b, 64*s + n, m]) * W2[64*s + n, l]
// D(16x16) = A(16x32, f16) x B(32x16, f16) + C(f32), two K-steps cover n=64.
// One wave handles one (b, 16-wide m-tile); 8 waves per 256-thread block.
// ---------------------------------------------------------------------------
__global__ __launch_bounds__(256) void qkv_wmma_kernel(
    const float* __restrict__ user, const float* __restrict__ item,
    const float* __restrict__ att,  const float* __restrict__ W1,
    const float* __restrict__ W2,
    float* __restrict__ Qf, float* __restrict__ Kf, float* __restrict__ Vf)
{
    const int tid    = threadIdx.x;
    const int lane   = tid & 31;
    const int wave   = tid >> 5;                 // 0..7
    const int lane16 = lane & 15;
    const int half   = lane >> 4;
    const int mt     = blockIdx.x * 8 + wave;    // 0..127  (m tile)
    const int b      = blockIdx.y;               // 0..15
    const int m      = mt * 16 + lane16;         // this lane's raw column

    v8f acc[3][2];
    const v8f zero = {};
    for (int s = 0; s < 3; ++s)
        for (int lt = 0; lt < 2; ++lt)
            acc[s][lt] = zero;

    for (int s = 0; s < 3; ++s) {                // Q / K / V segment
        const int nseg = 64 * s;
        for (int kt = 0; kt < 2; ++kt) {         // two K=32 steps cover n=64
            const int nbase = nseg + 32 * kt;

            // B operand (32x16): lanes 0-15 hold K=0-15, lanes 16-31 K=16-31,
            // column N = lane16, element j -> K = j + 16*half.
            v16h bmat;
            for (int j = 0; j < 16; ++j) {
                const int n = nbase + j + 16 * half;
                float v = raw_val(user, item, att, W1, b, n, m);
                v = v > 0.f ? v : 0.f;           // relu
                bmat[j] = (_Float16)v;
            }

            for (int lt = 0; lt < 2; ++lt) {     // two 16-row tiles of l (L=32)
                const int l = lt * 16 + lane16;  // A row M = lane16
                // A operand (16x32): VGPR i holds K-pair; K=2i+16*half,
                // matching B's contiguous K order.
                v16h amat;
                for (int i = 0; i < 8; ++i) {
                    const int kb = 2 * i + 16 * half;
                    const int n0 = nbase + kb;
                    amat[2 * i]     = (_Float16)W2[n0 * LL + l];        // W2^T
                    amat[2 * i + 1] = (_Float16)W2[(n0 + 1) * LL + l];
                }
                acc[s][lt] = __builtin_amdgcn_wmma_f32_16x16x32_f16(
                    false, amat, false, bmat, (short)0, acc[s][lt], false, false);
            }
        }
    }

    // C/D layout: VGPR r -> row M = r + 8*half, col N = lane16.
    float* const outs[3] = { Qf, Kf, Vf };
    for (int s = 0; s < 3; ++s)
        for (int lt = 0; lt < 2; ++lt)
            for (int r = 0; r < 8; ++r) {
                const int l = lt * 16 + r + 8 * half;
                outs[s][((size_t)b * LL + l) * MM + m] = acc[s][lt][r];
            }
}

// ---------------------------------------------------------------------------
// Kernel 2: the two needed columns of `causal`.
//   h1 = (Q[:,m].K[:,n0]) * adj[m,n0]*iw[m,n0]
//   h2 = (Q[:,n0].K[:,m]) * adj[n0,m]*iw[n0,m]
//   caus[t][b][m] = exp(min(d,0)) / (exp(min(d,0))+exp(min(-d,0))) * adj[m,n0]
// ---------------------------------------------------------------------------
__global__ __launch_bounds__(256) void causal_cols_kernel(
    const float* __restrict__ Qf, const float* __restrict__ Kf,
    const float* __restrict__ adj, const float* __restrict__ iw,
    float* __restrict__ caus)
{
    const int m  = blockIdx.x * 256 + threadIdx.x;  // 0..2047
    const int b  = blockIdx.y;
    const int t  = blockIdx.z;                      // 0 -> n0=0, 1 -> n0=2047
    const int n0 = t ? (MM - 1) : 0;

    const float* Qb = Qf + (size_t)b * LL * MM;
    const float* Kb = Kf + (size_t)b * LL * MM;

    float s1 = 0.f, s2 = 0.f;
    for (int l = 0; l < LL; ++l) {
        const float qm = Qb[l * MM + m];
        const float km = Kb[l * MM + m];
        const float qn = Qb[l * MM + n0];
        const float kn = Kb[l * MM + n0];
        s1 = fmaf(qm, kn, s1);
        s2 = fmaf(qn, km, s2);
    }
    const float a_mn = adj[(size_t)m * MM + n0];
    const float h1 = s1 * a_mn * iw[(size_t)m * MM + n0];
    const float h2 = s2 * adj[(size_t)n0 * MM + m] * iw[(size_t)n0 * MM + m];
    const float d  = h1 - h2;
    const float e1 = __expf(d > 0.f ? 0.f : d);
    const float e2 = __expf(-d > 0.f ? 0.f : -d);
    caus[((size_t)t * BB + b) * MM + m] = e1 / (e1 + e2) * a_mn;
}

// ---------------------------------------------------------------------------
// Kernel 3: out[b,l,t] = sum_m Value[b,l,m] * caus[t][b][m]
// One 256-thread block per output element; LDS tree reduction.
// ---------------------------------------------------------------------------
__global__ __launch_bounds__(256) void ans_reduce_kernel(
    const float* __restrict__ Vf, const float* __restrict__ caus,
    float* __restrict__ out)
{
    __shared__ float red[256];
    const int tid = threadIdx.x;
    const int l   = blockIdx.x;   // 0..31
    const int b   = blockIdx.y;   // 0..15
    const int t   = blockIdx.z;   // 0..1

    const float* Vrow = Vf   + ((size_t)b * LL + l) * MM;
    const float* crow = caus + ((size_t)t * BB + b) * MM;

    float s = 0.f;
    for (int m = tid; m < MM; m += 256)
        s = fmaf(Vrow[m], crow[m], s);
    red[tid] = s;
    __syncthreads();
    for (int off = 128; off > 0; off >>= 1) {
        if (tid < off) red[tid] += red[tid + off];
        __syncthreads();
    }
    if (tid == 0)
        out[((size_t)b * LL + l) * 2 + t] = red[0];
}

// ---------------------------------------------------------------------------
extern "C" void kernel_launch(void* const* d_in, const int* in_sizes, int n_in,
                              void* d_out, int out_size, void* d_ws, size_t ws_size,
                              hipStream_t stream)
{
    const float* user = (const float*)d_in[0];   // (B, 192, 1)
    const float* item = (const float*)d_in[1];   // (B, 192, 1)
    const float* att  = (const float*)d_in[2];   // (B, 192, 2046)
    const float* adj  = (const float*)d_in[3];   // (2048, 2048)
    const float* iw   = (const float*)d_in[4];   // (2048, 2048)
    const float* W1   = (const float*)d_in[5];   // (192, 2046)
    const float* W2   = (const float*)d_in[6];   // (192, 32)
    float* out = (float*)d_out;                  // (B, 32, 2)

    // Workspace: Q/K/V (B*32*2048 f32 each) + 2 causal columns sets.
    float* Qf   = (float*)d_ws;
    float* Kf   = Qf + (size_t)BB * LL * MM;
    float* Vf   = Kf + (size_t)BB * LL * MM;
    float* caus = Vf + (size_t)BB * LL * MM;     // [2][B][2048]
    // total = 3*1,048,576 + 65,536 floats = 12.25 MiB

    dim3 g1(16, BB, 1);     // 16 blocks * 8 waves = 128 m-tiles, per batch
    qkv_wmma_kernel<<<g1, 256, 0, stream>>>(user, item, att, W1, W2, Qf, Kf, Vf);

    dim3 g2(MM / 256, BB, 2);
    causal_cols_kernel<<<g2, 256, 0, stream>>>(Qf, Kf, adj, iw, caus);

    dim3 g3(LL, BB, 2);
    ans_reduce_kernel<<<g3, 256, 0, stream>>>(Vf, caus, out);
}
